// EstimatorNetwork_67087389164230
// MI455X (gfx1250) — matrix operand order repaired and checked by reading, not compiled, not run on hardware
//
#include <hip/hip_runtime.h>
#include <hip/hip_bf16.h>

// ---------------------------------------------------------------------------
// EstimatorNetwork on MI455X (gfx1250, wave32, WMMA)
//
// total[b] = sum over f of row-sums of the masked error chain
//   err_f = mask_f( bias_f + err_{f-1} @ W_{f-1}^T )
//
// 31 dependent GEMMs of (1024 x 2304) x (2304 x 2304)^T, 337 GFLOP total.
// Weights stream once from HBM (658 MB -> ~28us floor @ 23.3 TB/s); each W_f
// (21.2 MB) fits the 192 MB L2 so M-tile re-reads hit L2. Math on
// v_wmma_f32_16x16x32_f16 (f32 accumulate); fp32 weights converted to f16 in
// the LDS staging step. LDS is double-buffered: the f16 activation tile is
// staged with GLOBAL_LOAD_ASYNC_TO_LDS_B128 (ASYNCcnt) and the weight tile's
// fp32 loads are issued before the WMMA block, so global latency overlaps
// matrix math.
// ---------------------------------------------------------------------------

#define NUM_BLOCKS_PER_ROW 64
constexpr int N = 64 * 36;   // 2304
constexpr int F = 32;
constexpr int B = 1024;
constexpr int S = 256;

typedef __attribute__((ext_vector_type(16))) _Float16 v16h;
typedef __attribute__((ext_vector_type(8)))  _Float16 v8h;
typedef __attribute__((ext_vector_type(8)))  float    v8f;

// ---------------------------------------------------------------------------
// gfx1250 async copy: global -> LDS, 16 bytes per lane, tracked by ASYNCcnt.
// INST_OFFSET is added to BOTH the global and LDS address (ISA 08 §4.4).
// ---------------------------------------------------------------------------
__device__ __forceinline__ void async_g2l_b128(uint32_t lds_addr, uint64_t gaddr) {
    asm volatile("global_load_async_to_lds_b128 %0, %1, off"
                 :: "v"(lds_addr), "v"(gaddr) : "memory");
}
__device__ __forceinline__ void async_g2l_b128_off16(uint32_t lds_addr, uint64_t gaddr) {
    asm volatile("global_load_async_to_lds_b128 %0, %1, off offset:16"
                 :: "v"(lds_addr), "v"(gaddr) : "memory");
}
__device__ __forceinline__ void wait_asynccnt0() {
    asm volatile("s_wait_asynccnt 0x0" ::: "memory");
}
__device__ __forceinline__ uint32_t lds_offset_of(const void* p) {
    // Generic LDS pointers carry the offset in addr[31:0] (ISA 00 §aperture).
    return (uint32_t)(uintptr_t)p;
}

// ---------------------------------------------------------------------------
// Prep kernels
// ---------------------------------------------------------------------------
__global__ void init_keep_kernel(float* __restrict__ keep) {
    int i = blockIdx.x * blockDim.x + threadIdx.x;
    if (i < F * N) keep[i] = 1.0f;
}

__global__ void scatter_sel_kernel(const int* __restrict__ sel,
                                   float* __restrict__ keep) {
    int s = threadIdx.x;
    if (s < S) {
        int f    = sel[3 * s + 0];
        int node = sel[3 * s + 1] * NUM_BLOCKS_PER_ROW + sel[3 * s + 2];
        keep[f * N + node] = 0.0f;
    }
}

__global__ void init_err0_kernel(const float* __restrict__ bias0,
                                 const float* __restrict__ keep0,
                                 const int* __restrict__ cand,
                                 _Float16* __restrict__ err,
                                 float* __restrict__ out) {
    int b = blockIdx.x * blockDim.x + threadIdx.x;
    if (b >= B) return;
    int cf = cand[3 * b + 0];
    int cn = cand[3 * b + 1] * NUM_BLOCKS_PER_ROW + cand[3 * b + 2];
    float tot = 0.0f;
    for (int n = 0; n < N; ++n) {
        float v = bias0[n] * keep0[n];
        if (cf == 0 && cn == n) v = 0.0f;
        err[(size_t)b * N + n] = (_Float16)v;
        tot += v;
    }
    out[b] = tot;   // initializes poisoned d_out deterministically
}

// ---------------------------------------------------------------------------
// One GEMM step:  cur = mask_f(bias_f + prev @ W^T),  out[b] += rowsum(cur)
// Block = 128 threads (4 waves), 64x64 output tile; each wave owns a 32x32
// quadrant (2x2 WMMA subtiles). K-chunk = 32, double-buffered LDS pipeline.
// ---------------------------------------------------------------------------
__global__ __launch_bounds__(128)
void gemm_step_kernel(const float* __restrict__ Wf,      // (N, N) row-major
                      const float* __restrict__ bias_f,  // (N)
                      const float* __restrict__ keep_f,  // (N)
                      const int* __restrict__ cand,      // (B, 3)
                      const _Float16* __restrict__ prev, // (B, N)
                      _Float16* __restrict__ cur,        // (B, N)
                      float* __restrict__ out,           // (B)
                      int f) {
    __shared__ __align__(32) _Float16 sA[2][64 * 32];  // prev tile (rows x K)
    __shared__ __align__(32) _Float16 sW[2][64 * 32];  // W tile (out-cols x K)
    __shared__ int sCf[64];
    __shared__ int sCn[64];

    const int tid  = threadIdx.x;
    const int lane = tid & 31;
    const int wave = tid >> 5;           // 0..3
    const int wr   = wave >> 1;          // wave row   (0..1): rows wr*32..+31
    const int wc   = wave & 1;           // wave col   (0..1): cols wc*32..+31
    const int half = (lane >> 4) & 1;    // wave32 half-select
    const int c    = lane & 15;

    const int m0 = blockIdx.y * 64;      // output row base
    const int n0 = blockIdx.x * 64;      // output col base

    const int srow = tid >> 1;           // staging row this thread owns
    const int scg  = (tid & 1) * 16;     // staging col group (16 elements)

    if (tid < 64) {
        int b = m0 + tid;
        sCf[tid] = cand[3 * b + 0];
        sCn[tid] = cand[3 * b + 1] * NUM_BLOCKS_PER_ROW + cand[3 * b + 2];
    }

    // ---- prologue: stage chunk 0 into buffer 0 --------------------------
    {
        const _Float16* ga = prev + (size_t)(m0 + srow) * N + 0 + scg;
        uint32_t la = lds_offset_of(&sA[0][srow * 32 + scg]);
        async_g2l_b128(la, (uint64_t)(uintptr_t)ga);
        async_g2l_b128_off16(la, (uint64_t)(uintptr_t)ga);

        const float4* gw = (const float4*)(Wf + (size_t)(n0 + srow) * N + 0 + scg);
        float4 w0 = gw[0], w1 = gw[1], w2 = gw[2], w3 = gw[3];
        _Float16* dw = &sW[0][srow * 32 + scg];
        dw[0]  = (_Float16)w0.x; dw[1]  = (_Float16)w0.y; dw[2]  = (_Float16)w0.z; dw[3]  = (_Float16)w0.w;
        dw[4]  = (_Float16)w1.x; dw[5]  = (_Float16)w1.y; dw[6]  = (_Float16)w1.z; dw[7]  = (_Float16)w1.w;
        dw[8]  = (_Float16)w2.x; dw[9]  = (_Float16)w2.y; dw[10] = (_Float16)w2.z; dw[11] = (_Float16)w2.w;
        dw[12] = (_Float16)w3.x; dw[13] = (_Float16)w3.y; dw[14] = (_Float16)w3.z; dw[15] = (_Float16)w3.w;
    }
    wait_asynccnt0();
    __syncthreads();

    v8f acc[2][2] = {};          // [m-subtile][n-subtile] 16x16 f32 each
    constexpr int ITERS = N / 32;   // 72

    for (int k = 0; k < ITERS; ++k) {
        const int kb   = k & 1;
        const bool more = (k + 1 < ITERS);
        const int kn0  = (k + 1) * 32;

        // ---- issue next chunk's loads (overlap with WMMA below) ---------
        float4 w0, w1, w2, w3;
        if (more) {
            const _Float16* ga = prev + (size_t)(m0 + srow) * N + kn0 + scg;
            uint32_t la = lds_offset_of(&sA[kb ^ 1][srow * 32 + scg]);
            async_g2l_b128(la, (uint64_t)(uintptr_t)ga);
            async_g2l_b128_off16(la, (uint64_t)(uintptr_t)ga);

            const float4* gw = (const float4*)(Wf + (size_t)(n0 + srow) * N + kn0 + scg);
            w0 = gw[0]; w1 = gw[1]; w2 = gw[2]; w3 = gw[3];
        }
        // speculative prefetch two chunks ahead (global_prefetch_b8)
        if (k + 2 < ITERS && tid < 64) {
            __builtin_prefetch(Wf + (size_t)(n0 + tid) * N + kn0 + 32, 0, 1);
        }

        // ---- compute on buffer kb --------------------------------------
        const _Float16* Ab = sA[kb];
        const _Float16* Wb = sW[kb];

        // B fragments (32x16): lane -> column c, element e -> K = half*16+e
        v16h bfrag[2];
#pragma unroll
        for (int ns = 0; ns < 2; ++ns)
            bfrag[ns] = *(const v16h*)(Wb + (wc * 32 + ns * 16 + c) * 32 + half * 16);

#pragma unroll
        for (int ms = 0; ms < 2; ++ms) {
            // A fragment (16x32): lane -> row c,
            // element e -> K = half*16+e
            union { v16h v; v8h h[2]; } af;
            const _Float16* ar = Ab + (wr * 32 + ms * 16 + c) * 32;
            af.h[0] = *(const v8h*)(ar + half * 16);
            af.h[1] = *(const v8h*)(ar + half * 16 + 8);
#pragma unroll
            for (int ns = 0; ns < 2; ++ns) {
                acc[ms][ns] = __builtin_amdgcn_wmma_f32_16x16x32_f16(
                    false, af.v, false, bfrag[ns],
                    (short)0, acc[ms][ns], false, false);
            }
        }

        // ---- finish staging next chunk (cvt + LDS store after WMMAs) ---
        if (more) {
            _Float16* dw = &sW[kb ^ 1][srow * 32 + scg];
            dw[0]  = (_Float16)w0.x; dw[1]  = (_Float16)w0.y; dw[2]  = (_Float16)w0.z; dw[3]  = (_Float16)w0.w;
            dw[4]  = (_Float16)w1.x; dw[5]  = (_Float16)w1.y; dw[6]  = (_Float16)w1.z; dw[7]  = (_Float16)w1.w;
            dw[8]  = (_Float16)w2.x; dw[9]  = (_Float16)w2.y; dw[10] = (_Float16)w2.z; dw[11] = (_Float16)w2.w;
            dw[12] = (_Float16)w3.x; dw[13] = (_Float16)w3.y; dw[14] = (_Float16)w3.z; dw[15] = (_Float16)w3.w;
        }
        wait_asynccnt0();       // A tile for k+1 resident
        __syncthreads();        // W stores visible; buffer kb free for k+2
    }

    // ---- epilogue: bias + masks, store f16 cur, row-sum reduction -------
    float bn[2], kn[2];
#pragma unroll
    for (int ns = 0; ns < 2; ++ns) {
        int nglob = n0 + wc * 32 + ns * 16 + c;
        bn[ns] = bias_f[nglob];
        kn[ns] = keep_f[nglob];
    }

#pragma unroll
    for (int ms = 0; ms < 2; ++ms) {
#pragma unroll
        for (int v = 0; v < 8; ++v) {
            const int lr = wr * 32 + ms * 16 + half * 8 + v;  // local row
            const int b  = m0 + lr;                           // global row
            float rowpart = 0.0f;
#pragma unroll
            for (int ns = 0; ns < 2; ++ns) {
                const int nglob = n0 + wc * 32 + ns * 16 + c;
                float val = (acc[ms][ns][v] + bn[ns]) * kn[ns];
                if (sCf[lr] == f && sCn[lr] == nglob) val = 0.0f;
                cur[(size_t)b * N + nglob] = (_Float16)val;
                rowpart += val;
            }
            // sum the 16 column lanes (xor<16 stays within wave32 half)
            rowpart += __shfl_xor(rowpart, 1, 32);
            rowpart += __shfl_xor(rowpart, 2, 32);
            rowpart += __shfl_xor(rowpart, 4, 32);
            rowpart += __shfl_xor(rowpart, 8, 32);
            if (c == 0) atomicAdd(out + b, rowpart);
        }
    }
}

// ---------------------------------------------------------------------------
// Host-side launch (graph-capture safe: only kernel launches on `stream`)
// ---------------------------------------------------------------------------
extern "C" void kernel_launch(void* const* d_in, const int* in_sizes, int n_in,
                              void* d_out, int out_size, void* d_ws, size_t ws_size,
                              hipStream_t stream) {
    const float* weights = (const float*)d_in[0];   // (F-1, N, N)
    const float* biases  = (const float*)d_in[1];   // (F, N)
    const int*   sel     = (const int*)d_in[2];     // (S, 3)
    const int*   cand    = (const int*)d_in[3];     // (B, 3)
    float*       out     = (float*)d_out;           // (B,)

    char* ws = (char*)d_ws;
    float*    keep = (float*)ws;                              // F*N floats
    _Float16* buf0 = (_Float16*)(ws + (size_t)F * N * sizeof(float));
    _Float16* buf1 = buf0 + (size_t)B * N;

    init_keep_kernel<<<(F * N + 255) / 256, 256, 0, stream>>>(keep);
    scatter_sel_kernel<<<1, S, 0, stream>>>(sel, keep);
    init_err0_kernel<<<B / 256, 256, 0, stream>>>(biases, keep, cand, buf0, out);

    dim3 grid(N / 64, B / 64);   // 36 x 16 tiles
    for (int f = 1; f < F; ++f) {
        const _Float16* prev = (f & 1) ? buf0 : buf1;
        _Float16*       curb = (f & 1) ? buf1 : buf0;
        gemm_step_kernel<<<grid, 128, 0, stream>>>(
            weights + (size_t)(f - 1) * N * N,
            biases + (size_t)f * N,
            keep + (size_t)f * N,
            cand, prev, curb, out, f);
    }
}
